// Recurrent_block_74131135529625
// MI455X (gfx1250) — matrix operand order repaired and checked by reading, not compiled, not run on hardware
//
#include <hip/hip_runtime.h>
#include <hip/hip_bf16.h>

// ---------------------------------------------------------------------------
// Recurrent block (Griffin RG-LRU style) for MI455X / gfx1250.
// All six 16384x1024x1024 GEMMs via v_wmma_f32_16x16x32_bf16 (f32 accum).
// ---------------------------------------------------------------------------

typedef __attribute__((ext_vector_type(16))) __bf16 v16bf;
typedef __attribute__((ext_vector_type(8)))  float  v8f;
typedef __attribute__((ext_vector_type(4)))  unsigned int v4u;

#define BDIM 4
#define SLEN 4096
#define DDIM 1024
#define HDIM 1024
#define MROWS (BDIM * SLEN)          // 16384
#define CHUNK 64
#define NCHUNK (SLEN / CHUNK)        // 64

// ---------------- scalar helpers ----------------
__device__ __forceinline__ unsigned short f2bf(float f) {
    union { float f; unsigned int u; } c; c.f = f;
    unsigned int u = c.u;
    return (unsigned short)((u + 0x7FFFu + ((u >> 16) & 1u)) >> 16); // RNE
}
__device__ __forceinline__ float bf2f(unsigned short h) {
    union { unsigned int u; float f; } c; c.u = ((unsigned int)h) << 16;
    return c.f;
}
__device__ __forceinline__ float sigmoidf_(float x) {
    return 1.0f / (1.0f + __expf(-x));
}

// ---------------- fragment loads (documented CDNA5 16-bit layouts) ----------
__device__ __forceinline__ v16bf make_frag(v4u lo, v4u hi) {
    union U { v4u u2[2]; v16bf f; } x;
    x.u2[0] = lo; x.u2[1] = hi;
    return x.f;
}

// A fragment 16x32 (MxK) from row-major [*, ld] bf16.
// lane<16  holds row (lane),    K = {0..7, 16..23}
// lane>=16 holds row (lane-16), K = {8..15, 24..31}
__device__ __forceinline__ v16bf load_a_frag(const __bf16* __restrict__ tile,
                                             int ld, int lane) {
    int r = lane & 15;
    int h = lane >> 4;
    const __bf16* p = tile + r * ld + h * 8;
    v4u lo = *reinterpret_cast<const v4u*>(p);        // 8 bf16
    v4u hi = *reinterpret_cast<const v4u*>(p + 16);   // 8 bf16
    return make_frag(lo, hi);
}

// B fragment 32x16 (KxN) sourced from weight W[N,K] row-major (Bwmma = W^T):
// lane L holds column (L&15) = weight row (n0 + (L&15)),
// K = {(L>>4)*8 .. +7, 16+(L>>4)*8 .. +7} in that weight row.
__device__ __forceinline__ v16bf load_b_frag(const __bf16* __restrict__ wtile,
                                             int ld, int lane) {
    int c = lane & 15;
    int h = lane >> 4;
    const __bf16* p = wtile + c * ld + h * 8;
    v4u lo = *reinterpret_cast<const v4u*>(p);
    v4u hi = *reinterpret_cast<const v4u*>(p + 16);
    return make_frag(lo, hi);
}

// ---------------- f32 -> bf16 conversion ----------------
__global__ __launch_bounds__(256) void cvt_f32_bf16(const float* __restrict__ in,
                                                    unsigned short* __restrict__ out,
                                                    int n) {
    int i = blockIdx.x * 256 + threadIdx.x;
    if (i < n) out[i] = f2bf(in[i]);
}

// ---------------- WMMA GEMM:  C[M,N] = act(A[M,K] * W[N,K]^T + bias) -------
// Block = 256 threads = 8 waves. Wave w: rows [by*128 + w*16, +16),
// cols [bx*64, +64) as 4 WMMA n-tiles. K looped in steps of 32.
enum { EP_GELU_F32 = 0, EP_BIAS_BF16 = 1, EP_RELU_BF16 = 2, EP_BIAS_F32 = 3 };

template <int MODE>
__global__ __launch_bounds__(256) void gemm_bf16_wmma(
    const __bf16* __restrict__ A,   // [M,K] bf16
    const __bf16* __restrict__ W,   // [N,K] bf16
    const float*  __restrict__ bias,// [N]
    float*        __restrict__ Cf,  // f32 out (modes 0,3)
    unsigned short* __restrict__ Cb,// bf16 out (modes 1,2)
    int M, int N, int K) {
    const int lane = threadIdx.x & 31;
    const int wid  = threadIdx.x >> 5;
    const int m0   = blockIdx.y * 128 + wid * 16;
    const int n0   = blockIdx.x * 64;

    v8f acc0 = {}, acc1 = {}, acc2 = {}, acc3 = {};
    const __bf16* Arow = A + (size_t)m0 * K;
    const __bf16* W0 = W + (size_t)(n0 +  0) * K;
    const __bf16* W1 = W + (size_t)(n0 + 16) * K;
    const __bf16* W2 = W + (size_t)(n0 + 32) * K;
    const __bf16* W3 = W + (size_t)(n0 + 48) * K;

    for (int k = 0; k < K; k += 32) {
        v16bf af = load_a_frag(Arow + k, K, lane);
        v16bf b0 = load_b_frag(W0 + k, K, lane);
        v16bf b1 = load_b_frag(W1 + k, K, lane);
        v16bf b2 = load_b_frag(W2 + k, K, lane);
        v16bf b3 = load_b_frag(W3 + k, K, lane);
        acc0 = __builtin_amdgcn_wmma_f32_16x16x32_bf16(false, af, false, b0,
                                                       (short)0, acc0, false, false);
        acc1 = __builtin_amdgcn_wmma_f32_16x16x32_bf16(false, af, false, b1,
                                                       (short)0, acc1, false, false);
        acc2 = __builtin_amdgcn_wmma_f32_16x16x32_bf16(false, af, false, b2,
                                                       (short)0, acc2, false, false);
        acc3 = __builtin_amdgcn_wmma_f32_16x16x32_bf16(false, af, false, b3,
                                                       (short)0, acc3, false, false);
    }

    // C/D layout: lane<16 -> N=lane,   VGPR i -> M = i
    //             lane>=16-> N=lane-16,VGPR i -> M = 8+i
    const int nn = lane & 15;
    const int mb = (lane >> 4) * 8;
    v8f accs[4] = {acc0, acc1, acc2, acc3};
#pragma unroll
    for (int t = 0; t < 4; ++t) {
        const int col = n0 + t * 16 + nn;
        const float bv = bias[col];
#pragma unroll
        for (int i = 0; i < 8; ++i) {
            const int row = m0 + mb + i;
            float v = accs[t][i] + bv;
            if (MODE == EP_GELU_F32) {
                v = 0.5f * v * (1.0f + erff(v * 0.70710678118654752f));
                Cf[(size_t)row * N + col] = v;
            } else if (MODE == EP_BIAS_BF16) {
                Cb[(size_t)row * N + col] = f2bf(v);
            } else if (MODE == EP_RELU_BF16) {
                Cb[(size_t)row * N + col] = f2bf(fmaxf(v, 0.0f));
            } else { // EP_BIAS_F32
                Cf[(size_t)row * N + col] = v;
            }
        }
    }
}

// ---------------- RG-LRU gate prep (in-place rpre->a, ipre->g) -------------
__global__ __launch_bounds__(256) void gates_kernel(
    float* __restrict__ rpre_a,          // in: r_pre, out: a
    float* __restrict__ ipre_g,          // in: i_pre, out: g
    const unsigned short* __restrict__ xcb,
    const float* __restrict__ Lam, int total) {
    int i = blockIdx.x * 256 + threadIdx.x;
    if (i >= total) return;
    int hc = i & (HDIM - 1);
    float sp = log1pf(__expf(-Lam[hc]));          // softplus(-Lam)
    float r  = sigmoidf_(rpre_a[i]);
    float ig = sigmoidf_(ipre_g[i]);
    float a  = __expf(-8.0f * sp * r);            // C = 8
    float xc = bf2f(xcb[i]);
    float g  = sqrtf(fmaxf(0.0f, 1.0f - a * a)) * (ig * xc);
    rpre_a[i] = a;
    ipre_g[i] = g;
}

// ---------------- 3-pass chunked linear scan: h = a*h + g ------------------
// layout of P/Q/Hs: [b][chunk][h]
__global__ __launch_bounds__(256) void scan_pass1(
    const float* __restrict__ a, const float* __restrict__ g,
    float* __restrict__ P, float* __restrict__ Q) {
    int tid = blockIdx.x * 256 + threadIdx.x;          // B*NCHUNK*HDIM
    int h = tid & (HDIM - 1);
    int c = (tid >> 10) & (NCHUNK - 1);
    int b = tid >> 16;
    size_t base = ((size_t)b * SLEN + (size_t)c * CHUNK) * HDIM + h;
    float Pv = 1.0f, Qv = 0.0f;
    for (int s = 0; s < CHUNK; ++s) {
        float av = a[base + (size_t)s * HDIM];
        float gv = g[base + (size_t)s * HDIM];
        Qv = av * Qv + gv;
        Pv = av * Pv;
    }
    P[tid] = Pv;
    Q[tid] = Qv;
}

__global__ __launch_bounds__(256) void scan_pass2(
    const float* __restrict__ P, const float* __restrict__ Q,
    float* __restrict__ Hs) {
    int tid = blockIdx.x * 256 + threadIdx.x;          // B*HDIM
    int h = tid & (HDIM - 1);
    int b = tid >> 10;
    float run = 0.0f;
    for (int c = 0; c < NCHUNK; ++c) {
        int idx = (b * NCHUNK + c) * HDIM + h;
        Hs[idx] = run;                                  // state entering chunk c
        run = P[idx] * run + Q[idx];
    }
}

__global__ __launch_bounds__(256) void scan_pass3(
    const float* __restrict__ a, const float* __restrict__ g,
    const float* __restrict__ x1, const float* __restrict__ Hs,
    unsigned short* __restrict__ yb) {
    int tid = blockIdx.x * 256 + threadIdx.x;          // B*NCHUNK*HDIM
    int h = tid & (HDIM - 1);
    int c = (tid >> 10) & (NCHUNK - 1);
    int b = tid >> 16;
    size_t base = ((size_t)b * SLEN + (size_t)c * CHUNK) * HDIM + h;
    float hs = Hs[tid];
    for (int s = 0; s < CHUNK; ++s) {
        size_t idx = base + (size_t)s * HDIM;
        hs = a[idx] * hs + g[idx];
        yb[idx] = f2bf(x1[idx] * hs);                  // gated merge x1 * x2
    }
}

// ---------------------------------------------------------------------------
extern "C" void kernel_launch(void* const* d_in, const int* in_sizes, int n_in,
                              void* d_out, int out_size, void* d_ws, size_t ws_size,
                              hipStream_t stream) {
    // inputs (setup_inputs order)
    const float* x      = (const float*)d_in[0];   // [B,S,D]
    const float* Wp1    = (const float*)d_in[1];   // [H,D]
    const float* bp1    = (const float*)d_in[2];
    const float* Wp2    = (const float*)d_in[3];
    const float* bp2    = (const float*)d_in[4];
    const float* conv_w = (const float*)d_in[5];   // [TCO,H]
    const float* conv_b = (const float*)d_in[6];
    const float* Wa     = (const float*)d_in[7];
    const float* ba     = (const float*)d_in[8];
    const float* Wx     = (const float*)d_in[9];
    const float* bx     = (const float*)d_in[10];
    const float* Lam    = (const float*)d_in[11];
    const float* Wp3    = (const float*)d_in[12];  // [D,H]
    const float* bp3    = (const float*)d_in[13];
    float* out = (float*)d_out;                    // [B,S,D] f32

    // ---- workspace carving (bytes) ----
    const size_t ACT_F32 = (size_t)MROWS * HDIM * sizeof(float);          // 64 MB
    const size_t ACT_BF  = (size_t)MROWS * HDIM * sizeof(unsigned short); // 32 MB
    const size_t WGT_BF  = (size_t)HDIM * DDIM * sizeof(unsigned short);  //  2 MB
    char* ws = (char*)d_ws;
    size_t off = 0;
    float* X1   = (float*)(ws + off); off += ACT_F32;     // gelu branch
    float* Abuf = (float*)(ws + off); off += ACT_F32;     // r_pre -> a
    float* Gbuf = (float*)(ws + off); off += ACT_F32;     // i_pre -> g
    unsigned short* xb  = (unsigned short*)(ws + off); off += ACT_BF;  // x bf16
    unsigned short* x2b = (unsigned short*)(ws + off); off += ACT_BF;  // branch2 lin
    unsigned short* xcb = (unsigned short*)(ws + off); off += ACT_BF;  // conv+relu
    unsigned short* w1b = (unsigned short*)(ws + off); off += WGT_BF;
    unsigned short* w2b = (unsigned short*)(ws + off); off += WGT_BF;
    unsigned short* wcb = (unsigned short*)(ws + off); off += WGT_BF;
    unsigned short* wab = (unsigned short*)(ws + off); off += WGT_BF;
    unsigned short* wxb = (unsigned short*)(ws + off); off += WGT_BF;
    unsigned short* w3b = (unsigned short*)(ws + off); off += WGT_BF;
    float* Pbuf = (float*)(ws + off); off += (size_t)BDIM * NCHUNK * HDIM * 4;
    float* Qbuf = (float*)(ws + off); off += (size_t)BDIM * NCHUNK * HDIM * 4;
    float* Hsb  = (float*)(ws + off); off += (size_t)BDIM * NCHUNK * HDIM * 4;
    unsigned short* yb = x2b;   // x2b dead after conv GEMM -> reuse for y bf16

    const int NACT = MROWS * HDIM;    // 16,777,216
    const int NWGT = HDIM * DDIM;     //  1,048,576

    // ---- 1) f32 -> bf16 conversions ----
    cvt_f32_bf16<<<NACT / 256, 256, 0, stream>>>(x, xb, NACT);
    cvt_f32_bf16<<<NWGT / 256, 256, 0, stream>>>(Wp1,    w1b, NWGT);
    cvt_f32_bf16<<<NWGT / 256, 256, 0, stream>>>(Wp2,    w2b, NWGT);
    cvt_f32_bf16<<<NWGT / 256, 256, 0, stream>>>(conv_w, wcb, NWGT);
    cvt_f32_bf16<<<NWGT / 256, 256, 0, stream>>>(Wa,     wab, NWGT);
    cvt_f32_bf16<<<NWGT / 256, 256, 0, stream>>>(Wx,     wxb, NWGT);
    cvt_f32_bf16<<<NWGT / 256, 256, 0, stream>>>(Wp3,    w3b, NWGT);

    dim3 ggrid(HDIM / 64, MROWS / 128);   // (16, 128)
    dim3 gblk(256);

    // ---- 2) branch 1: x1 = gelu(x * Wp1^T + bp1) ----
    gemm_bf16_wmma<EP_GELU_F32><<<ggrid, gblk, 0, stream>>>(
        (const __bf16*)xb, (const __bf16*)w1b, bp1, X1, nullptr,
        MROWS, HDIM, DDIM);
    // ---- 3) branch 2 linear: x2 = x * Wp2^T + bp2  (bf16 out) ----
    gemm_bf16_wmma<EP_BIAS_BF16><<<ggrid, gblk, 0, stream>>>(
        (const __bf16*)xb, (const __bf16*)w2b, bp2, nullptr, x2b,
        MROWS, HDIM, DDIM);
    // ---- 4) pointwise conv (k=1) + ReLU: xc = relu(x2 * conv_w^T + conv_b) ----
    gemm_bf16_wmma<EP_RELU_BF16><<<ggrid, gblk, 0, stream>>>(
        (const __bf16*)x2b, (const __bf16*)wcb, conv_b, nullptr, xcb,
        MROWS, HDIM, HDIM);
    // ---- 5) gate GEMMs: r_pre, i_pre ----
    gemm_bf16_wmma<EP_BIAS_F32><<<ggrid, gblk, 0, stream>>>(
        (const __bf16*)xcb, (const __bf16*)wab, ba, Abuf, nullptr,
        MROWS, HDIM, HDIM);
    gemm_bf16_wmma<EP_BIAS_F32><<<ggrid, gblk, 0, stream>>>(
        (const __bf16*)xcb, (const __bf16*)wxb, bx, Gbuf, nullptr,
        MROWS, HDIM, HDIM);
    // ---- 6) elementwise gates: a, g (in place) ----
    gates_kernel<<<NACT / 256, 256, 0, stream>>>(Abuf, Gbuf, xcb, Lam, NACT);
    // ---- 7) chunked linear recurrence + gated merge -> yb (bf16) ----
    const int NP = BDIM * NCHUNK * HDIM;                // 262,144
    scan_pass1<<<NP / 256, 256, 0, stream>>>(Abuf, Gbuf, Pbuf, Qbuf);
    scan_pass2<<<(BDIM * HDIM) / 256, 256, 0, stream>>>(Pbuf, Qbuf, Hsb);
    scan_pass3<<<NP / 256, 256, 0, stream>>>(Abuf, Gbuf, X1, Hsb, yb);
    // ---- 8) output projection: out = y * Wp3^T + bp3 ----
    gemm_bf16_wmma<EP_BIAS_F32><<<ggrid, gblk, 0, stream>>>(
        (const __bf16*)yb, (const __bf16*)w3b, bp3, out, nullptr,
        MROWS, DDIM, HDIM);

    (void)in_sizes; (void)n_in; (void)out_size; (void)ws_size;
}
